// WordsCapsLayer_11184094839435
// MI455X (gfx1250) — matrix operand order repaired and checked by reading, not compiled, not run on hardware
//
#include <hip/hip_runtime.h>

#define BATCH    64
#define IN_CAPS  512
#define IN_DIM   128
#define NUM_CAPS 32
#define DIM_CAPS 32

typedef __attribute__((ext_vector_type(16))) __bf16 v16bf;
typedef __attribute__((ext_vector_type(8)))  float  v8f;

// ---------------------------------------------------------------------------
// Pack X (B,IN_CAPS,IN_DIM f32) into hi/lo bf16 arrays laid out in the exact
// WMMA A-operand order, so the GEMM loads one v16bf per lane contiguously.
// A (16-bit 16x32): lane = 16*h + m15; elements 0..7 -> K = 8h+e,
// elements 8..15 -> K = 16+8h+e.  Packed: [b*32+mtile][kstep][lane][e]
// ---------------------------------------------------------------------------
__global__ __launch_bounds__(256) void pack_A(const float* __restrict__ X,
                                              __bf16* __restrict__ Ahi,
                                              __bf16* __restrict__ Alo) {
  const size_t t = (size_t)blockIdx.x * 256 + threadIdx.x; // b*512*128 + row*128 + d
  const int    d   = (int)(t & (IN_DIM - 1));
  const size_t bi  = t >> 7;
  const int    row = (int)(bi & (IN_CAPS - 1));
  const int    b   = (int)(bi >> 9);

  const float  x  = X[t];
  const __bf16 hi = (__bf16)x;
  const __bf16 lo = (__bf16)(x - (float)hi);

  const int m15 = row & 15, mt = row >> 4;
  const int ks  = d >> 5,   r  = d & 31;
  const int h   = (r >> 3) & 1;
  const int e   = (r & 7) | ((r >> 4) << 3);
  const size_t dst = ((((size_t)(b * 32 + mt)) * 4 + ks) * 32 + (h * 16 + m15)) * 16 + e;
  Ahi[dst] = hi;
  Alo[dst] = lo;
}

// ---------------------------------------------------------------------------
// Pack W (NUM_CAPS,IN_DIM,DIM_CAPS f32) into hi/lo bf16 in WMMA B-operand
// order. B (32x16 KxN): lane = 16*bh + n15; elements 0..7 -> K = 8bh+e, 8..15 -> K = 16+8bh+e.
// Packed: [n][j][kstep][lane][e]   (512 KB total, L2 resident)
// ---------------------------------------------------------------------------
__global__ __launch_bounds__(256) void pack_B(const float* __restrict__ W,
                                              __bf16* __restrict__ Bhi,
                                              __bf16* __restrict__ Blo) {
  const int t  = blockIdx.x * 256 + threadIdx.x;  // n*128*32 + d*32 + c
  const int c  = t & 31;
  const int dn = t >> 5;
  const int d  = dn & (IN_DIM - 1);
  const int n  = dn >> 7;

  const float  w  = W[t];
  const __bf16 hi = (__bf16)w;
  const __bf16 lo = (__bf16)(w - (float)hi);

  const int kk = d >> 5, r = d & 31;
  const int bh = (r >> 3) & 1, be = (r & 7) | ((r >> 4) << 3);
  const int j  = c >> 4;
  const int dst = ((((n * 2 + j) * 4 + kk) * 32) + (bh * 16 + (c & 15))) * 16 + be;
  Bhi[dst] = hi;
  Blo[dst] = lo;
}

// ---------------------------------------------------------------------------
// u_hat[b][n][i][c] = sum_d X[b][i][d] * W[n][d][c] via bf16 hi/lo split
// (3x V_WMMA_F32_16X16X32_BF16 per 32-K step, ~f32 accuracy).
// Block = 128 thr (4 waves); wave owns one 16-row M tile x 32 cols x 4 n's.
// Grid = B * (IN_CAPS/64) * (NUM_CAPS/4) = 4096 blocks.
// Per K-step: 24 WMMA vs 18 v16bf loads (2 A, 16 B); A L2 traffic amortized 4x over n.
// ---------------------------------------------------------------------------
__global__ __launch_bounds__(128) void uhat_gemm(const __bf16* __restrict__ Ahi,
                                                 const __bf16* __restrict__ Alo,
                                                 const __bf16* __restrict__ Bhi,
                                                 const __bf16* __restrict__ Blo,
                                                 float* __restrict__ uhat) {
  const int blk    = blockIdx.x;          // b(64) x mchunk(8) x ngroup(8)
  const int ng     = blk & 7;
  const int mchunk = (blk >> 3) & 7;
  const int b      = blk >> 6;

  const int lane = threadIdx.x & 31;
  const int wave = threadIdx.x >> 5;
  const int mt   = mchunk * 4 + wave;     // M tile 0..31
  const int n0   = ng * 4;

  v8f acc[4][2] = {};

  const size_t abase = (((size_t)(b * 32 + mt)) * 4 * 32 + lane) * 16;
#pragma unroll
  for (int kk = 0; kk < 4; ++kk) {
    const v16bf a_hi = *(const v16bf*)(Ahi + abase + (size_t)kk * 512);
    const v16bf a_lo = *(const v16bf*)(Alo + abase + (size_t)kk * 512);
#pragma unroll
    for (int nn = 0; nn < 4; ++nn) {
#pragma unroll
      for (int j = 0; j < 2; ++j) {
        const size_t bofs = ((((size_t)(n0 + nn) * 2 + j) * 4 + kk) * 32 + lane) * 16;
        const v16bf b_hi = *(const v16bf*)(Bhi + bofs);
        const v16bf b_lo = *(const v16bf*)(Blo + bofs);
        acc[nn][j] = __builtin_amdgcn_wmma_f32_16x16x32_bf16(
            false, a_hi, false, b_hi, (short)0, acc[nn][j], false, false);
        acc[nn][j] = __builtin_amdgcn_wmma_f32_16x16x32_bf16(
            false, a_hi, false, b_lo, (short)0, acc[nn][j], false, false);
        acc[nn][j] = __builtin_amdgcn_wmma_f32_16x16x32_bf16(
            false, a_lo, false, b_hi, (short)0, acc[nn][j], false, false);
      }
    }
  }

  // C/D layout: lane half -> M base (0/8), VGPR v -> M = base+v, N = lane&15.
  const int l15  = lane & 15;
  const int h    = lane >> 4;
  const int mrow = mt * 16 + h * 8;
#pragma unroll
  for (int nn = 0; nn < 4; ++nn) {
    float* Un = uhat + ((size_t)(b * 32 + n0 + nn)) * IN_CAPS * DIM_CAPS;
#pragma unroll
    for (int v = 0; v < 8; ++v) {
      Un[(size_t)(mrow + v) * DIM_CAPS + l15]      = acc[nn][0][v];
      Un[(size_t)(mrow + v) * DIM_CAPS + 16 + l15] = acc[nn][1][v];
    }
  }
}

// ---------------------------------------------------------------------------
// softmax stats over n: per (b,i) compute max_n and sum_n exp(logit - max).
// ---------------------------------------------------------------------------
__global__ __launch_bounds__(256) void softmax_stats(const float* __restrict__ bLog,
                                                     float* __restrict__ mx,
                                                     float* __restrict__ se) {
  const int idx = blockIdx.x * 256 + threadIdx.x;  // b*512 + i
  const int b = idx >> 9;
  const int i = idx & 511;
  const float* p = bLog + (size_t)b * NUM_CAPS * IN_CAPS + i;
  float m = -3.4e38f;
#pragma unroll 4
  for (int n = 0; n < NUM_CAPS; ++n) m = fmaxf(m, p[n * IN_CAPS]);
  float s = 0.f;
#pragma unroll 4
  for (int n = 0; n < NUM_CAPS; ++n) s += __expf(p[n * IN_CAPS] - m);
  mx[idx] = m;
  se[idx] = s;
}

// ---------------------------------------------------------------------------
// One routing step per (b,n): s = sum_i c_i*uhat_i ; v = squash(s);
// FINAL ? out = v : logits += uhat . v   (uhat streams from L2)
// ---------------------------------------------------------------------------
template <bool FINAL>
__global__ __launch_bounds__(256) void route_iter(const float* __restrict__ uhat,
                                                  float* __restrict__ bLog,
                                                  const float* __restrict__ mx,
                                                  const float* __restrict__ se,
                                                  float* __restrict__ out) {
  const int bn  = blockIdx.x;     // b*32 + n
  const int b   = bn >> 5;
  const int tid = threadIdx.x;
  const int c   = tid & 31;
  const int ig  = tid >> 5;       // 0..7

  const float* U   = uhat + (size_t)bn * IN_CAPS * DIM_CAPS;
  const float* bl  = bLog + (size_t)bn * IN_CAPS;
  const float* mxb = mx + (size_t)b * IN_CAPS;
  const float* seb = se + (size_t)b * IN_CAPS;

  __shared__ float spart[8][32];
  __shared__ float vsh[32];

  float acc = 0.f;
  for (int i = ig; i < IN_CAPS; i += 8) {
    const float cc = __expf(bl[i] - mxb[i]) / seb[i];
    acc += cc * U[(size_t)i * DIM_CAPS + c];
  }
  spart[ig][c] = acc;
  __syncthreads();

  if (tid < 32) {
    float s = 0.f;
#pragma unroll
    for (int g = 0; g < 8; ++g) s += spart[g][tid];
    spart[0][tid] = s;
  }
  __syncthreads();

  if (tid < 32) {
    float sn = 0.f;
#pragma unroll
    for (int k = 0; k < 32; ++k) { const float t = spart[0][k]; sn += t * t; }
    const float sv = spart[0][tid];
    const float v  = (sn / (1.f + sn)) * sv / (sqrtf(sn) + 1e-8f);
    if (FINAL) out[(size_t)bn * DIM_CAPS + tid] = v;
    vsh[tid] = v;
  }
  __syncthreads();

  if (!FINAL) {
    float* blw = bLog + (size_t)bn * IN_CAPS;
    for (int i = tid; i < IN_CAPS; i += 256) {
      const float4* ur = (const float4*)(U + (size_t)i * DIM_CAPS);
      float dot = 0.f;
#pragma unroll
      for (int q = 0; q < 8; ++q) {
        const float4 u4 = ur[q];
        dot += u4.x * vsh[q * 4 + 0] + u4.y * vsh[q * 4 + 1] +
               u4.z * vsh[q * 4 + 2] + u4.w * vsh[q * 4 + 3];
      }
      blw[i] += dot;
    }
  }
}

// ---------------------------------------------------------------------------
extern "C" void kernel_launch(void* const* d_in, const int* in_sizes, int n_in,
                              void* d_out, int out_size, void* d_ws, size_t ws_size,
                              hipStream_t stream) {
  const float* X  = (const float*)d_in[0];   // (64,512,128)
  const float* W  = (const float*)d_in[1];   // (32,128,32)
  const float* b0 = (const float*)d_in[2];   // (64,32,512,1)
  float* out = (float*)d_out;                // (64,32,32)

  const size_t UHAT_ELEMS = (size_t)BATCH * NUM_CAPS * IN_CAPS * DIM_CAPS; // 33.5M f32
  const size_t BLOG_ELEMS = (size_t)BATCH * NUM_CAPS * IN_CAPS;            // 1M f32
  const size_t STAT_ELEMS = (size_t)BATCH * IN_CAPS;                       // 32K f32
  const size_t APK_ELEMS  = (size_t)BATCH * IN_CAPS * IN_DIM;              // 4.2M bf16
  const size_t BPK_ELEMS  = (size_t)NUM_CAPS * IN_DIM * DIM_CAPS;          // 131K bf16

  char* ws = (char*)d_ws;
  float*  uhat = (float*)ws;                       ws += UHAT_ELEMS * 4;
  float*  bLog = (float*)ws;                       ws += BLOG_ELEMS * 4;
  float*  mx   = (float*)ws;                       ws += STAT_ELEMS * 4;
  float*  se   = (float*)ws;                       ws += STAT_ELEMS * 4;
  __bf16* Ahi  = (__bf16*)ws;                      ws += APK_ELEMS * 2;
  __bf16* Alo  = (__bf16*)ws;                      ws += APK_ELEMS * 2;
  __bf16* Bhi  = (__bf16*)ws;                      ws += BPK_ELEMS * 2;
  __bf16* Blo  = (__bf16*)ws;

  // 0) one-time hi/lo bf16 split + WMMA-operand swizzle of X and W
  pack_A<<<(int)(APK_ELEMS / 256), 256, 0, stream>>>(X, Ahi, Alo);
  pack_B<<<(int)(BPK_ELEMS / 256), 256, 0, stream>>>(W, Bhi, Blo);

  // 1) u_hat via WMMA (134 MB output stays L2-resident for routing)
  uhat_gemm<<<BATCH * (IN_CAPS / 64) * (NUM_CAPS / 4), 128, 0, stream>>>(
      Ahi, Alo, Bhi, Blo, uhat);

  // 2) logits = b0 (inputs stay immutable)
  hipMemcpyAsync(bLog, b0, BLOG_ELEMS * sizeof(float),
                 hipMemcpyDeviceToDevice, stream);

  const int statGrid = (BATCH * IN_CAPS) / 256;

  // 3) two logit-update routing iterations
  for (int it = 0; it < 2; ++it) {
    softmax_stats<<<statGrid, 256, 0, stream>>>(bLog, mx, se);
    route_iter<false><<<BATCH * NUM_CAPS, 256, 0, stream>>>(uhat, bLog, mx, se, nullptr);
  }

  // 4) final softmax + squash -> output
  softmax_stats<<<statGrid, 256, 0, stream>>>(bLog, mx, se);
  route_iter<true><<<BATCH * NUM_CAPS, 256, 0, stream>>>(uhat, bLog, mx, se, out);
}
